// KernelAttention_34703335752275
// MI455X (gfx1250) — matrix operand order repaired and checked by reading, not compiled, not run on hardware
//
#include <hip/hip_runtime.h>
#include <hip/hip_bf16.h>

// Problem constants (from reference): x[16,256,64,64], W[128,256], b[128],
// P[512,128], gamma scalar. T = 65536 tokens, D = 128, K = 512.

typedef float  v8f  __attribute__((ext_vector_type(8)));
typedef __bf16 v16bf __attribute__((ext_vector_type(16)));

union FragAB { unsigned u[8]; v16bf v; };

__device__ __forceinline__ unsigned short f2bf_rne(float f) {
    unsigned u = __float_as_uint(f);
    u += 0x7FFFu + ((u >> 16) & 1u);           // round-to-nearest-even
    return (unsigned short)(u >> 16);
}
__device__ __forceinline__ unsigned packbf(float lo, float hi) {
    return (unsigned)f2bf_rne(lo) | ((unsigned)f2bf_rne(hi) << 16);
}
// float pair -> packed bf16 dword via native casts (lets compiler pick v_cvt_pk)
__device__ __forceinline__ unsigned pack2bf(float lo, float hi) {
    union { __bf16 h[2]; unsigned u; } t;
    t.h[0] = (__bf16)lo; t.h[1] = (__bf16)hi;
    return t.u;
}

// ---------------------------------------------------------------------------
// Prep: pack W (128x256, row=d) and P (512x128, row=k) into bf16 B-operand
// layout (K-pairs packed into u32: one 32-bit load per B-fragment VGPR).
// Wpk[p][d] = (W[d][2p], W[d][2p+1]); Ppk[p][k] = (P[k][2p], P[k][2p+1]).
// Also p_norm2[k] in f32.
// ---------------------------------------------------------------------------
__global__ void proto_prep_kernel(const float* __restrict__ W,
                                  const float* __restrict__ P,
                                  unsigned* __restrict__ Wpk,
                                  unsigned* __restrict__ Ppk,
                                  float* __restrict__ pn) {
    int i = blockIdx.x * 256 + threadIdx.x;
    if (i < 128 * 128) {                        // Wpk: p in [0,128), d in [0,128)
        int p = i >> 7, d = i & 127;
        Wpk[i] = packbf(W[d * 256 + 2 * p], W[d * 256 + 2 * p + 1]);
    } else if (i < 16384 + 32768) {             // Ppk: p in [0,64), k in [0,512)
        int j = i - 16384;
        int p = j >> 9, k = j & 511;
        Ppk[j] = packbf(P[k * 128 + 2 * p], P[k * 128 + 2 * p + 1]);
    } else if (i < 16384 + 32768 + 512) {       // p_norm2
        int k = i - 49152;
        float s = 0.f;
        for (int d = 0; d < 128; ++d) { float v = P[k * 128 + d]; s += v * v; }
        pn[k] = s;
    }
}

// ---------------------------------------------------------------------------
// Fused kernel: per block = 64 tokens (never crosses image n: 4096 % 64 == 0).
//   LDS: ls[64][520] f32 logits (overlays xf[256][68] f32 x-tile, async-filled),
//        qs[64][130] bf16 q-tile, rscale[64].
//   8 waves: wave w -> token strip mt = w>>1 (16 rows), column half w&1.
// ---------------------------------------------------------------------------
#define LS_STRIDE 520
#define XF_STRIDE 68                                // f32; 272B rows (16B aligned)
#define QS_STRIDE 130
#define QS_OFF    (64 * LS_STRIDE * 4)              // 133120
#define RS_OFF    (QS_OFF + 64 * QS_STRIDE * 2)     // +16640
#define SHM_BYTES (RS_OFF + 64 * 4)                 // 150016

__global__ void __launch_bounds__(256)
proto_softmax_fused_kernel(const float* __restrict__ x,
                           const float* __restrict__ bias,
                           const float* __restrict__ gamma,
                           const unsigned* __restrict__ Wpk,
                           const unsigned* __restrict__ Ppk,
                           const float* __restrict__ pn,
                           float* __restrict__ out) {
    extern __shared__ char smem[];
    float*          ls = (float*)smem;                       // [64][520]
    float*          xf = (float*)smem;                       // [256][68] (overlay)
    unsigned short* qs = (unsigned short*)(smem + QS_OFF);   // [64][130]
    float*      rscale = (float*)(smem + RS_OFF);            // [64]

    const int tid = threadIdx.x;
    const int t0  = blockIdx.x * 64;
    const int n   = t0 >> 12;            // token / 4096
    const int s0  = t0 & 4095;           // h*64+w base
    const float gab = fabsf(gamma[0]);

    // ---- Stage x tile (f32) into LDS via async DMA: xf[c][m], 16B chunks --
    // Global: xb[c*4096 + m] (contiguous over m). 4096 quads / 256 threads.
    const float* xb = x + ((size_t)n << 20) + s0;
    {
        // low 32 bits of a generic LDS pointer == wave-relative LDS byte addr
        unsigned xfBase = (unsigned)(unsigned long long)(void*)xf;
        #pragma unroll
        for (int it = 0; it < 16; ++it) {
            int i  = it * 256 + tid;
            int c  = i >> 4;
            int mq = (i & 15) << 2;
            unsigned goff = (unsigned)(c * 4096 + mq) * 4u;          // bytes
            unsigned loff = xfBase + (unsigned)(c * XF_STRIDE + mq) * 4u;
            asm volatile("global_load_async_to_lds_b128 %0, %1, %2"
                         :: "v"(loff), "v"(goff), "s"(xb) : "memory");
        }
    }
#if __has_builtin(__builtin_amdgcn_s_wait_asynccnt)
    __builtin_amdgcn_s_wait_asynccnt(0);
#else
    asm volatile("s_wait_asynccnt 0" ::: "memory");
#endif
    __syncthreads();

    const int w = tid >> 5, lane = tid & 31;
    const int g = lane >> 4, ml = lane & 15;
    const int mt = w >> 1;               // 16-token strip
    const int nh = w & 1;                // column half
    const int m0 = mt * 16 + ml;         // A-matrix row owned by this lane

    // ---- GEMM1: q[64x128] = Xbf16 @ Wbf16^T, K=256 in 8 steps of 32 -------
    v8f accq[4];
    #pragma unroll
    for (int t = 0; t < 4; ++t) { v8f z = {0,0,0,0,0,0,0,0}; accq[t] = z; }

    for (int ks = 0; ks < 8; ++ks) {
        FragAB a;                        // A frag: K = 16g + 2j (contiguous), cvt on read
        #pragma unroll
        for (int j = 0; j < 8; ++j) {
            int k0 = ks * 32 + 16 * g + 2 * j;
            a.u[j] = pack2bf(xf[k0 * XF_STRIDE + m0],
                             xf[(k0 + 1) * XF_STRIDE + m0]);
        }
        #pragma unroll
        for (int t = 0; t < 4; ++t) {
            int col = (nh * 4 + t) * 16 + ml;
            FragAB bm;                   // B frag: K = 16g + 2j (pair-packed)
            #pragma unroll
            for (int j = 0; j < 8; ++j)
                bm.u[j] = Wpk[(size_t)(ks * 16 + 8 * g + j) * 128 + col];
            accq[t] = __builtin_amdgcn_wmma_f32_16x16x32_bf16(
                false, a.v, false, bm.v, (short)0, accq[t], false, false);
        }
    }

    // ---- q + bias -> bf16 LDS [m][d] --------------------------------------
    #pragma unroll
    for (int t = 0; t < 4; ++t) {
        int d = (nh * 4 + t) * 16 + ml;
        float bv = bias[d];
        #pragma unroll
        for (int r = 0; r < 8; ++r) {
            int m = mt * 16 + 8 * g + r;             // C/D layout: M = r + 8g
            qs[m * QS_STRIDE + d] = f2bf_rne(accq[t][r] + bv);
        }
    }
    __syncthreads();   // xf reads done; ls region now writable

    // ---- GEMM2: logits[64x512] = q @ P^T, K=128 in 4 steps; 4 col-chunks --
    const unsigned short* qRow = qs + m0 * QS_STRIDE;
    for (int cc = 0; cc < 4; ++cc) {
        v8f accl[4];
        #pragma unroll
        for (int t = 0; t < 4; ++t) { v8f z = {0,0,0,0,0,0,0,0}; accl[t] = z; }

        for (int ks = 0; ks < 4; ++ks) {
            FragAB a;
            #pragma unroll
            for (int j = 0; j < 8; ++j) {
                int k0 = ks * 32 + 16 * g + 2 * j;
                a.u[j] = *(const unsigned*)(qRow + k0);
            }
            #pragma unroll
            for (int t = 0; t < 4; ++t) {
                int kc = ((nh * 16 + cc * 4 + t) * 16) + ml;
                FragAB bm;
                #pragma unroll
                for (int j = 0; j < 8; ++j)
                    bm.u[j] = Ppk[(size_t)(ks * 16 + 8 * g + j) * 512 + kc];
                accl[t] = __builtin_amdgcn_wmma_f32_16x16x32_bf16(
                    false, a.v, false, bm.v, (short)0, accl[t], false, false);
            }
        }
        // logit = 2|g|*(q.p) - |g|*pn[k]   (q_norm2 cancels in softmax)
        #pragma unroll
        for (int t = 0; t < 4; ++t) {
            int kc = ((nh * 16 + cc * 4 + t) * 16) + ml;
            float pk = pn[kc];
            #pragma unroll
            for (int r = 0; r < 8; ++r) {
                int m = mt * 16 + 8 * g + r;
                ls[m * LS_STRIDE + kc] = 2.0f * gab * accl[t][r] - gab * pk;
            }
        }
    }
    __syncthreads();

    // ---- Softmax over K=512: 4 lanes per token (quad-shuffle reduce) ------
    {
        int m = tid >> 2, part = tid & 3;
        float* lrow = ls + m * LS_STRIDE;
        float mx = -3.4e38f;
        for (int i = 0; i < 128; ++i) mx = fmaxf(mx, lrow[part + 4 * i]);
        mx = fmaxf(mx, __shfl_xor(mx, 1, 32));
        mx = fmaxf(mx, __shfl_xor(mx, 2, 32));
        float sum = 0.f;
        for (int i = 0; i < 128; ++i) {
            int k = part + 4 * i;
            float e = __expf(lrow[k] - mx);
            lrow[k] = e;
            sum += e;
        }
        sum += __shfl_xor(sum, 1, 32);
        sum += __shfl_xor(sum, 2, 32);
        if (part == 0) rscale[m] = 1.0f / sum;
    }
    __syncthreads();

    // ---- Coalesced b128 store: out[n,k,s0+mq..mq+3] -----------------------
    float* ob = out + ((size_t)n << 21) + s0;
    for (int i = tid; i < 512 * 16; i += 256) {       // 8192 quads
        int k  = i >> 4;
        int mq = (i & 15) << 2;
        float4 v;
        v.x = ls[(mq + 0) * LS_STRIDE + k] * rscale[mq + 0];
        v.y = ls[(mq + 1) * LS_STRIDE + k] * rscale[mq + 1];
        v.z = ls[(mq + 2) * LS_STRIDE + k] * rscale[mq + 2];
        v.w = ls[(mq + 3) * LS_STRIDE + k] * rscale[mq + 3];
        *(float4*)(ob + (size_t)k * 4096 + mq) = v;
    }
}

// ---------------------------------------------------------------------------
extern "C" void kernel_launch(void* const* d_in, const int* in_sizes, int n_in,
                              void* d_out, int out_size, void* d_ws, size_t ws_size,
                              hipStream_t stream) {
    const float* x     = (const float*)d_in[0];   // [16,256,64,64]
    const float* W     = (const float*)d_in[1];   // [128,256]
    const float* b     = (const float*)d_in[2];   // [128]
    const float* P     = (const float*)d_in[3];   // [512,128]
    const float* gamma = (const float*)d_in[4];   // scalar
    float* out = (float*)d_out;                   // [16,512,64,64]

    unsigned* Wpk = (unsigned*)d_ws;                          //  64 KB
    unsigned* Ppk = (unsigned*)((char*)d_ws + 65536);         // 128 KB
    float*    pn  = (float*)((char*)d_ws + 196608);           //   2 KB

    proto_prep_kernel<<<194, 256, 0, stream>>>(W, P, Wpk, Ppk, pn);

    hipFuncSetAttribute((const void*)proto_softmax_fused_kernel,
                        hipFuncAttributeMaxDynamicSharedMemorySize, SHM_BYTES);
    proto_softmax_fused_kernel<<<65536 / 64, 256, SHM_BYTES, stream>>>(
        x, b, gamma, Wpk, Ppk, pn, out);
}
